// GATNet_70007966924828
// MI455X (gfx1250) — compile-verified, matrix operand order checked by reading, not run on hardware
//
#include <hip/hip_runtime.h>
#include <hip/hip_bf16.h>

typedef __attribute__((ext_vector_type(16))) __bf16 v16bf;
typedef __attribute__((ext_vector_type(8)))  float  v8f;

#define N_NODES 100000
#define N_EDGES 1600000
#define IN_FEAT 128
#define HID 32
#define HEADS 4
#define N_CLASSES 16

// ---------------- atomic helpers ----------------
__device__ __forceinline__ void atomAddF(float* p, float v) {
    unsafeAtomicAdd(p, v);   // native global_atomic_add_f32
}

// float atomic max via int punning (works across sign mix, init = -inf)
__device__ __forceinline__ void atomMaxF(float* addr, float val) {
    if (!(val < 0.0f)) atomicMax((int*)addr, __float_as_int(val));
    else               atomicMin((unsigned int*)addr, (unsigned int)__float_as_int(val));
}

// ---------------- generic fill ----------------
__global__ void fill_f32(float* __restrict__ p, float v, int n) {
    int t = blockIdx.x * blockDim.x + threadIdx.x;
    if (t < n) p[t] = v;
}

// ---------------- weight pre-swizzle into WMMA B-fragment order (bf16) ------
// frag id f = j*4 + kk (j = 16-col tile, kk = K/32 step). Element e = lane*16+i.
// lane<16 : n = j*16+lane,   k = kk*32 + (i<8 ?  i   : i+8 )
// lane>=16: n = j*16+lane-16, k = kk*32 + (i<8 ? 8+i : i+16)
__global__ void prep_w(const float* __restrict__ W, __bf16* __restrict__ out,
                       int ncols, int nfrag) {
    int t = blockIdx.x * blockDim.x + threadIdx.x;
    if (t >= nfrag * 512) return;
    int f    = t >> 9;
    int e    = t & 511;
    int lane = e >> 4;
    int i    = e & 15;
    int j    = f >> 2;
    int kk   = f & 3;
    int l16  = lane >> 4;
    int n    = j * 16 + (lane & 15);
    int k    = (i < 8) ? (kk * 32 + (l16 ? 8 + i : i))
                       : (kk * 32 + (l16 ? i + 16 : i + 8));
    out[t] = (__bf16)W[k * ncols + n];
}

// ---------------- WMMA GEMM: out[N, CT*16] = X[N,128] @ W[128, CT*16] -------
template <int CT>
__global__ void gemm_wmma(const float* __restrict__ X,
                          const __bf16* __restrict__ Wb,
                          float* __restrict__ out, int rowTiles) {
    const int wave = threadIdx.x >> 5;
    const int lane = threadIdx.x & 31;
    const int rt   = blockIdx.x * (blockDim.x >> 5) + wave;
    if (rt >= rowTiles) return;            // wave-uniform: EXEC stays all-ones
    const int l16 = lane >> 4;
    const int r   = lane & 15;
    const int row = rt * 16 + r;

    v8f acc[CT] = {};

    const float4* xp = (const float4*)(X + (size_t)row * 128);
#pragma unroll
    for (int kk = 0; kk < 4; ++kk) {
        const int b0 = kk * 32 + (l16 ? 8 : 16 * 0 + 8 * l16); // see below
        // explicit: run0 start, run1 start
        const int s0 = kk * 32 + (l16 ? 8 : 0);
        const int s1 = kk * 32 + (l16 ? 24 : 16);
        (void)b0;
        float4 f0 = xp[s0 >> 2];
        float4 f1 = xp[(s0 >> 2) + 1];
        float4 f2 = xp[s1 >> 2];
        float4 f3 = xp[(s1 >> 2) + 1];
        v16bf a;
        a[0] = (__bf16)f0.x;  a[1] = (__bf16)f0.y;  a[2]  = (__bf16)f0.z;  a[3]  = (__bf16)f0.w;
        a[4] = (__bf16)f1.x;  a[5] = (__bf16)f1.y;  a[6]  = (__bf16)f1.z;  a[7]  = (__bf16)f1.w;
        a[8] = (__bf16)f2.x;  a[9] = (__bf16)f2.y;  a[10] = (__bf16)f2.z;  a[11] = (__bf16)f2.w;
        a[12] = (__bf16)f3.x; a[13] = (__bf16)f3.y; a[14] = (__bf16)f3.z;  a[15] = (__bf16)f3.w;
#pragma unroll
        for (int j = 0; j < CT; ++j) {
            v16bf b = *(const v16bf*)(Wb + ((size_t)((j * 4 + kk) * 32 + lane)) * 16);
            acc[j] = __builtin_amdgcn_wmma_f32_16x16x32_bf16(
                false, a, false, b, (short)0, acc[j], false, false);
        }
    }

    const int ncols = CT * 16;
#pragma unroll
    for (int j = 0; j < CT; ++j) {
#pragma unroll
        for (int v = 0; v < 8; ++v) {
            int rowo = rt * 16 + v + l16 * 8;
            out[(size_t)rowo * ncols + j * 16 + r] = acc[j][v];
        }
    }
}

// ---------------- per-node attention dot products ----------------
__global__ void attn_dots(const float* __restrict__ feat,
                          const float* __restrict__ al, const float* __restrict__ ar,
                          float* __restrict__ el, float* __restrict__ er,
                          int n, int heads, int dim) {
    int t = blockIdx.x * blockDim.x + threadIdx.x;
    if (t >= n * heads) return;
    int node = t / heads, h = t - node * heads;
    const float* f = feat + (size_t)node * heads * dim + h * dim;
    const float* L = al + h * dim;
    const float* R = ar + h * dim;
    float sl = 0.f, sr = 0.f;
    for (int d = 0; d < dim; ++d) { float v = f[d]; sl += v * L[d]; sr += v * R[d]; }
    el[t] = sl;
    er[t] = sr;
}

__device__ __forceinline__ float edge_e(const float* __restrict__ el,
                                        const float* __restrict__ er,
                                        int s, int d, int h, int H) {
    float v = el[s * H + h] + er[d * H + h];
    return v > 0.f ? v : 0.2f * v;
}

// ---------------- edge pass 1: segment max ----------------
__global__ void edge_max_k(const int* __restrict__ src, const int* __restrict__ dst,
                           const float* __restrict__ el, const float* __restrict__ er,
                           float* __restrict__ m, int E, int H) {
    int t = blockIdx.x * blockDim.x + threadIdx.x;
    if (t >= E * H) return;
    int e = t / H, h = t - e * H;
    int s = src[e], d = dst[e];
    atomMaxF(&m[d * H + h], edge_e(el, er, s, d, h, H));
}

// ---------------- edge pass 2: segment sum of exp ----------------
__global__ void edge_sum_k(const int* __restrict__ src, const int* __restrict__ dst,
                           const float* __restrict__ el, const float* __restrict__ er,
                           const float* __restrict__ m, float* __restrict__ z,
                           int E, int H) {
    int t = blockIdx.x * blockDim.x + threadIdx.x;
    if (t >= E * H) return;
    int e = t / H, h = t - e * H;
    int s = src[e], d = dst[e];
    float ev = edge_e(el, er, s, d, h, H);
    atomAddF(&z[d * H + h], __expf(ev - m[d * H + h]));
}

// ---------------- edge pass 3: weighted message scatter-add ----------------
__global__ void edge_msg_k(const int* __restrict__ src, const int* __restrict__ dst,
                           const float* __restrict__ el, const float* __restrict__ er,
                           const float* __restrict__ m, const float* __restrict__ z,
                           const float* __restrict__ feat, float* __restrict__ rst,
                           int E, int H, int D) {
    int t = blockIdx.x * blockDim.x + threadIdx.x;
    if (t >= E * H) return;
    int e = t / H, h = t - e * H;
    int s = src[e], d = dst[e];
    float ev = edge_e(el, er, s, d, h, H);
    float a  = __expf(ev - m[d * H + h]) / z[d * H + h];
    const float* fs = feat + (size_t)s * H * D + h * D;
    float*       rd = rst  + (size_t)d * H * D + h * D;
    for (int k = 0; k < D; k += 4) {
        float4 f = *(const float4*)(fs + k);
        atomAddF(rd + k + 0, a * f.x);
        atomAddF(rd + k + 1, a * f.y);
        atomAddF(rd + k + 2, a * f.z);
        atomAddF(rd + k + 3, a * f.w);
    }
}

// ---------------- bias + ELU (in place) ----------------
__global__ void elu_bias(float* __restrict__ h, const float* __restrict__ b,
                         int n, int C) {
    int t = blockIdx.x * blockDim.x + threadIdx.x;
    if (t >= n * C) return;
    float v = h[t] + b[t % C];
    h[t] = v > 0.f ? v : (__expf(v) - 1.f);
}

// ---------------- final: out = rst2 + b2 (mean over 1 head == identity) -----
__global__ void final_out(const float* __restrict__ rst, const float* __restrict__ b,
                          float* __restrict__ out, int n, int C) {
    int t = blockIdx.x * blockDim.x + threadIdx.x;
    if (t >= n * C) return;
    out[t] = rst[t] + b[t % C];
}

// =====================================================================
extern "C" void kernel_launch(void* const* d_in, const int* in_sizes, int n_in,
                              void* d_out, int out_size, void* d_ws, size_t ws_size,
                              hipStream_t stream) {
    const float* x   = (const float*)d_in[0];
    const int*   src = (const int*)d_in[1];
    const int*   dst = (const int*)d_in[2];
    const float* W1  = (const float*)d_in[3];
    const float* al1 = (const float*)d_in[4];
    const float* ar1 = (const float*)d_in[5];
    const float* b1  = (const float*)d_in[6];
    const float* W2  = (const float*)d_in[7];
    const float* al2 = (const float*)d_in[8];
    const float* ar2 = (const float*)d_in[9];
    const float* b2  = (const float*)d_in[10];
    float* out = (float*)d_out;
    char* ws = (char*)d_ws;

    // workspace layout (all offsets 128B aligned)
    float* feat1 = (float*)(ws + 0);                       // [N,128]
    float* hbuf  = (float*)(ws + 51200000);                // rst1 -> elu in place [N,128]
    float* el1   = (float*)(ws + 102400000);               // [N,4]
    float* er1   = (float*)(ws + 104000000);
    float* m1    = (float*)(ws + 105600000);
    float* z1    = (float*)(ws + 107200000);
    float* feat2 = (float*)(ws + 108800000);               // [N,16]
    float* rst2  = (float*)(ws + 115200000);               // [N,16]
    float* el2   = (float*)(ws + 121600000);               // [N]
    float* er2   = (float*)(ws + 122000000);
    float* m2    = (float*)(ws + 122400000);
    float* z2    = (float*)(ws + 122800000);
    __bf16* wb1  = (__bf16*)(ws + 123200000);              // 8*4 frags * 512
    __bf16* wb2  = (__bf16*)(ws + 123232768);              // 1*4 frags * 512

    const int TPB = 256;
    const int rowTiles = N_NODES / 16;                     // 6250, exact
    const int gemmBlocks = (rowTiles + 7) / 8;

    // ---- weight swizzle to bf16 fragments ----
    prep_w<<<(32 * 512 + TPB - 1) / TPB, TPB, 0, stream>>>(W1, wb1, 128, 32);
    prep_w<<<(4  * 512 + TPB - 1) / TPB, TPB, 0, stream>>>(W2, wb2, 16, 4);

    // ---- layer 1 init ----
    fill_f32<<<(N_NODES * 128 + TPB - 1) / TPB, TPB, 0, stream>>>(hbuf, 0.f, N_NODES * 128);
    fill_f32<<<(N_NODES * HEADS + TPB - 1) / TPB, TPB, 0, stream>>>(m1, -INFINITY, N_NODES * HEADS);
    fill_f32<<<(N_NODES * HEADS + TPB - 1) / TPB, TPB, 0, stream>>>(z1, 0.f, N_NODES * HEADS);

    // ---- layer 1: GEMM + attention dots ----
    gemm_wmma<8><<<gemmBlocks, TPB, 0, stream>>>(x, wb1, feat1, rowTiles);
    attn_dots<<<(N_NODES * HEADS + TPB - 1) / TPB, TPB, 0, stream>>>(
        feat1, al1, ar1, el1, er1, N_NODES, HEADS, HID);

    // ---- layer 1 edge softmax + aggregate ----
    int eh1 = N_EDGES * HEADS;
    edge_max_k<<<(eh1 + TPB - 1) / TPB, TPB, 0, stream>>>(src, dst, el1, er1, m1, N_EDGES, HEADS);
    edge_sum_k<<<(eh1 + TPB - 1) / TPB, TPB, 0, stream>>>(src, dst, el1, er1, m1, z1, N_EDGES, HEADS);
    edge_msg_k<<<(eh1 + TPB - 1) / TPB, TPB, 0, stream>>>(src, dst, el1, er1, m1, z1,
                                                          feat1, hbuf, N_EDGES, HEADS, HID);
    elu_bias<<<(N_NODES * 128 + TPB - 1) / TPB, TPB, 0, stream>>>(hbuf, b1, N_NODES, 128);

    // ---- layer 2 init ----
    fill_f32<<<(N_NODES * N_CLASSES + TPB - 1) / TPB, TPB, 0, stream>>>(rst2, 0.f, N_NODES * N_CLASSES);
    fill_f32<<<(N_NODES + TPB - 1) / TPB, TPB, 0, stream>>>(m2, -INFINITY, N_NODES);
    fill_f32<<<(N_NODES + TPB - 1) / TPB, TPB, 0, stream>>>(z2, 0.f, N_NODES);

    // ---- layer 2: GEMM + attention dots ----
    gemm_wmma<1><<<gemmBlocks, TPB, 0, stream>>>(hbuf, wb2, feat2, rowTiles);
    attn_dots<<<(N_NODES + TPB - 1) / TPB, TPB, 0, stream>>>(
        feat2, al2, ar2, el2, er2, N_NODES, 1, N_CLASSES);

    // ---- layer 2 edge softmax + aggregate ----
    edge_max_k<<<(N_EDGES + TPB - 1) / TPB, TPB, 0, stream>>>(src, dst, el2, er2, m2, N_EDGES, 1);
    edge_sum_k<<<(N_EDGES + TPB - 1) / TPB, TPB, 0, stream>>>(src, dst, el2, er2, m2, z2, N_EDGES, 1);
    edge_msg_k<<<(N_EDGES + TPB - 1) / TPB, TPB, 0, stream>>>(src, dst, el2, er2, m2, z2,
                                                              feat2, rst2, N_EDGES, 1, N_CLASSES);

    // ---- final ----
    final_out<<<(N_NODES * N_CLASSES + TPB - 1) / TPB, TPB, 0, stream>>>(rst2, b2, out, N_NODES, N_CLASSES);
}
